// Detections_21345987461432
// MI455X (gfx1250) — matrix operand order repaired and checked by reading, not compiled, not run on hardware
//
#include <hip/hip_runtime.h>
#include <hip/hip_bf16.h>
#include <cstddef>

#define B_      16
#define NPRED   25200
#define CSTRIDE 85
#define NCLS_   80
#define KDET    300
#define KPAD    304
#define MTGT    50
#define CONF_T  0.8f
#define NMS_T   0.4f

// output float offsets (reference tuple order, flattened)
#define OPB 0
#define OPS (B_*KDET*4)            // 19200
#define OPL (OPS + B_*KDET)        // 24000
#define OPV (OPL + B_*KDET)        // 28800
#define OTB (OPV + B_*KDET)        // 33600
#define OTS (OTB + B_*MTGT*4)      // 36800
#define OTL (OTS + B_*MTGT)        // 37600
#define OTV (OTL + B_*MTGT)        // 38400

typedef __attribute__((ext_vector_type(16))) _Float16 v16h;
typedef __attribute__((ext_vector_type(8)))  float    v8f;
typedef __attribute__((ext_vector_type(4)))  unsigned int v4u;
typedef __attribute__((ext_vector_type(8)))  int      v8i_t;
typedef __attribute__((ext_vector_type(4)))  int      v4i_t;

__device__ __forceinline__ int imin_(int a, int b) { return a < b ? a : b; }

#define ROWS_PER_BLOCK 128

#if defined(__AMDGCN__) && __has_builtin(__builtin_amdgcn_tensor_load_to_lds) && __has_builtin(__builtin_amdgcn_s_wait_tensorcnt)
#define USE_TDM 1
#else
#define USE_TDM 0
#endif

// ---------------------------------------------------------------------------
// Kernel 1: stream preds via TDM -> LDS, compute score = obj * max(cls),
// threshold at CONF_T, write score array.
// ---------------------------------------------------------------------------
__global__ __launch_bounds__(ROWS_PER_BLOCK)
void score_kernel(const float* __restrict__ preds, float* __restrict__ scores)
{
    __shared__ __align__(16) float tile[ROWS_PER_BLOCK * CSTRIDE];
    const int b    = blockIdx.y;
    const int row0 = blockIdx.x * ROWS_PER_BLOCK;
    const int rows = imin_(ROWS_PER_BLOCK, NPRED - row0);
    const float* gbase = preds + ((size_t)b * NPRED + row0) * CSTRIDE;

#if USE_TDM
    {
        // one 1-D tensor DMA per wave: 32 rows x 340B, as 8-byte units
        const int wid   = __builtin_amdgcn_readfirstlane((int)(threadIdx.x >> 5));
        const int wrow0 = wid * 32;
        const int wrows = imin_(32, rows - wrow0);
        if (wrows > 0) {
            const unsigned nunits = (unsigned)(wrows * CSTRIDE * 4) >> 3; // 8B units (rows even -> exact)
            const unsigned long long ga =
                (unsigned long long)(const void*)(gbase + (size_t)wrow0 * CSTRIDE);
            const unsigned ldsoff =
                (unsigned)(unsigned long long)(const void*)&tile[wrow0 * CSTRIDE];
            // D# group 0: count=1 | lds_addr | global_addr | type=2
            v4u g0;
            g0[0] = 1u;
            g0[1] = ldsoff;
            g0[2] = (unsigned)ga;
            g0[3] = ((unsigned)(ga >> 32) & 0x1FFFFFFu) | 0x80000000u;
            // D# group 1: data_size=8B, tensor_dim0=tile_dim0=nunits, dim1=1, stride0=nunits
            v8i_t g1;
            g1[0] = (int)(3u << 16);          // data_size=3 (8B), mask=0
            g1[1] = (int)(nunits << 16);      // tensor_dim0 low16 (nunits<65536)
            g1[2] = (int)(1u << 16);          // tensor_dim0 hi=0 | tensor_dim1=1
            g1[3] = (int)(nunits << 16);      // tensor_dim1 hi=0 | tile_dim0=nunits
            g1[4] = 1;                        // tile_dim1=1, tile_dim2=0
            g1[5] = (int)nunits;              // tensor_dim0_stride low32
            g1[6] = 0;
            g1[7] = 0;
            v4i_t z4; z4[0]=0; z4[1]=0; z4[2]=0; z4[3]=0;
            v8i_t z8; z8[0]=0; z8[1]=0; z8[2]=0; z8[3]=0; z8[4]=0; z8[5]=0; z8[6]=0; z8[7]=0;
            // 6-arg variant (clang-23 / therock-10.0 headers)
            __builtin_amdgcn_tensor_load_to_lds(g0, g1, z4, z4, z8, 0);
            __builtin_amdgcn_s_wait_tensorcnt(0);
        }
    }
#else
    {
        const int nchunks = (rows * CSTRIDE) >> 2;   // float4 chunks; rows%4==0 -> exact
        const float4* g4 = (const float4*)gbase;
        float4* t4 = (float4*)tile;
        for (int c = threadIdx.x; c < nchunks; c += ROWS_PER_BLOCK) t4[c] = g4[c];
    }
#endif
    __syncthreads();

    const int t = threadIdx.x;
    if (t < rows) {
        const float* r = &tile[t * CSTRIDE];      // stride 85 words: odd -> bank-conflict-free
        const float obj = r[4];
        float best = r[5];
        #pragma unroll 4
        for (int c = 1; c < NCLS_; ++c) {
            float v = r[5 + c];
            if (v > best) best = v;
        }
        const float s = obj * best;
        scores[(size_t)b * NPRED + row0 + t] = (s > CONF_T) ? s : 0.0f;
    }
}

// ---------------------------------------------------------------------------
// Kernel 2: exact top-300 per batch = 300 x argmax (tie -> smallest index,
// matching jax.lax.top_k stable ordering). Destroys scores (rewritten by k1).
// ---------------------------------------------------------------------------
__global__ __launch_bounds__(1024)
void topk_kernel(float* __restrict__ scores,
                 int* __restrict__ top_idx, float* __restrict__ top_s)
{
    __shared__ float sv[1024];
    __shared__ int   si[1024];
    const int b = blockIdx.x;
    const int t = threadIdx.x;
    float* sc = scores + (size_t)b * NPRED;

    for (int k = 0; k < KDET; ++k) {
        float bv = -3.402823466e38f;
        int   bi = 0;
        for (int n = t; n < NPRED; n += 1024) {
            float v = sc[n];
            if (v > bv) { bv = v; bi = n; }       // ascending n: first max kept
        }
        sv[t] = bv; si[t] = bi;
        __syncthreads();
        for (int off = 512; off > 0; off >>= 1) {
            if (t < off) {
                float v2 = sv[t + off]; int i2 = si[t + off];
                if (v2 > sv[t] || (v2 == sv[t] && i2 < si[t])) { sv[t] = v2; si[t] = i2; }
            }
            __syncthreads();
        }
        if (t == 0) {
            int bi0 = si[0];
            top_idx[b * KDET + k] = bi0;
            top_s  [b * KDET + k] = sv[0];
            sc[bi0] = -3.402823466e38f;
        }
        __syncthreads();
    }
}

// ---------------------------------------------------------------------------
// Kernel 3: per-batch NMS. Same-label 304x304 mask via WMMA one-hot GEMM
// (exact in f16/f32), IoU suppression bitmask, wave-synchronous greedy loop.
// ---------------------------------------------------------------------------
__global__ __launch_bounds__(320)
void nms_kernel(const float* __restrict__ preds,
                const int* __restrict__ top_idx, const float* __restrict__ top_s,
                float* __restrict__ out)
{
    __shared__ float bx1[KPAD], by1[KPAD], bx2[KPAD], by2[KPAD], barea[KPAD], bsc[KPAD];
    __shared__ int   blab[KPAD];
    __shared__ unsigned sameBits[KPAD * 10];   // 304 rows x 320 cols
    __shared__ unsigned suppBits[KDET * 10];   // 300 rows x 320 cols (j>i baked in)
    __shared__ unsigned keepW[10];

    const int b = blockIdx.x;
    const int t = threadIdx.x;

    // Phase 1: gather detections, convert boxes, recompute argmax label
    for (int i = t; i < KPAD; i += 320) {
        float x1 = 0.f, y1 = 0.f, x2 = 0.f, y2 = 0.f, s = 0.f;
        int lab = -1;
        if (i < KDET) {
            const int idx = top_idx[b * KDET + i];
            s = top_s[b * KDET + i];
            const float* p = preds + ((size_t)b * NPRED + idx) * CSTRIDE;
            const float cx = p[0], cy = p[1], w = p[2], h = p[3];
            x1 = cx - 0.5f * w; y1 = cy - 0.5f * h; x2 = x1 + w; y2 = y1 + h;
            float best = p[5]; int bid = 0;
            for (int c = 1; c < NCLS_; ++c) {
                float v = p[5 + c];
                if (v > best) { best = v; bid = c; }
            }
            lab = bid;
        }
        bx1[i] = x1; by1[i] = y1; bx2[i] = x2; by2[i] = y2; bsc[i] = s; blab[i] = lab;
        barea[i] = fmaxf(x2 - x1, 0.f) * fmaxf(y2 - y1, 0.f);
    }
    for (int i = t; i < KPAD * 10; i += 320) sameBits[i] = 0u;
    __syncthreads();

    // Phase 2: same-label mask via WMMA: D = onehot(label) . onehot(label)^T
    {
        const int wv   = t >> 5;
        const int lane = t & 31;
        const _Float16 h1 = (_Float16)1.0f, h0 = (_Float16)0.0f;
        for (int tile = wv; tile < 19 * 19; tile += 10) {
            const int tm = tile / 19, tn = tile % 19;
            const int lm = blab[tm * 16 + (lane & 15)];   // A: row M
            const int ln = blab[tn * 16 + (lane & 15)];   // B: col N
            v8f acc = {};
            #pragma unroll
            for (int kc = 0; kc < 3; ++kc) {
                const int kb = kc * 32;
                v16h a, bb;
                #pragma unroll
                for (int e = 0; e < 16; ++e) {
                    const int kA = kb + ((lane < 16) ? ((e < 8) ? e : 16 + (e - 8))
                                                     : ((e < 8) ? 8 + e : 24 + (e - 8)));
                    const int kB = kA;
                    a[e]  = (lm == kA) ? h1 : h0;
                    bb[e] = (ln == kB) ? h1 : h0;
                }
                acc = __builtin_amdgcn_wmma_f32_16x16x32_f16(
                        false, a, false, bb, (short)0, acc, false, false);
            }
            const int wword = tn >> 1;
            const int shift = (tn & 1) * 16;
            #pragma unroll
            for (int r = 0; r < 8; ++r) {
                const unsigned mask = (unsigned)__ballot(acc[r] > 0.5f);
                if (lane == 0 && (mask & 0xFFFFu))
                    atomicOr(&sameBits[(tm * 16 + r) * 10 + wword], (mask & 0xFFFFu) << shift);
                if (lane == 16 && (mask >> 16))
                    atomicOr(&sameBits[(tm * 16 + 8 + r) * 10 + wword], (mask >> 16) << shift);
            }
        }
    }
    __syncthreads();

    // Phase 3: suppression bitmask (iou > thres && same label && j > i) + keep init
    for (int p = t; p < KDET * 10; p += 320) {
        const int i = p / 10, w = p % 10;
        const unsigned sameW = sameBits[i * 10 + w];
        unsigned m = 0;
        for (int jj = 0; jj < 32; ++jj) {
            const int j = w * 32 + jj;
            if (j < KDET && j > i && ((sameW >> jj) & 1u)) {
                const float xx1 = fmaxf(bx1[i], bx1[j]);
                const float yy1 = fmaxf(by1[i], by1[j]);
                const float xx2 = fminf(bx2[i], bx2[j]);
                const float yy2 = fminf(by2[i], by2[j]);
                const float inter = fmaxf(xx2 - xx1, 0.f) * fmaxf(yy2 - yy1, 0.f);
                const float iou = inter / (barea[i] + barea[j] - inter + 1e-9f);
                if (iou > NMS_T) m |= 1u << jj;
            }
        }
        suppBits[p] = m;
    }
    if (t < 10) {
        unsigned m = 0;
        for (int jj = 0; jj < 32; ++jj) {
            const int i = t * 32 + jj;
            if (i < KDET && bsc[i] > 0.0f) m |= 1u << jj;
        }
        keepW[t] = m;
    }
    __syncthreads();

    // Phase 4: sequential greedy NMS, wave-synchronous on wave 0 (LDS in-order)
    if (t < 32) {
        volatile unsigned* kw = keepW;
        for (int i = 0; i < KDET; ++i) {
            const unsigned w = kw[i >> 5];
            if ((w >> (i & 31)) & 1u) {
                if (t < 10) kw[t] = kw[t] & ~suppBits[i * 10 + t];
            }
        }
    }
    __syncthreads();

    // Phase 5: masked outputs
    for (int i = t; i < KDET; i += 320) {
        const bool kp = (keepW[i >> 5] >> (i & 31)) & 1u;
        const int g = b * KDET + i;
        out[OPB + g * 4 + 0] = kp ? bx1[i] : 0.f;
        out[OPB + g * 4 + 1] = kp ? by1[i] : 0.f;
        out[OPB + g * 4 + 2] = kp ? bx2[i] : 0.f;
        out[OPB + g * 4 + 3] = kp ? by2[i] : 0.f;
        out[OPS + g] = kp ? bsc[i] : 0.f;
        out[OPL + g] = kp ? (float)blab[i] : -1.f;
        out[OPV + g] = kp ? 1.f : 0.f;
    }
}

// ---------------------------------------------------------------------------
// Kernel 4: target reformatting (trivial)
// ---------------------------------------------------------------------------
__global__ void target_kernel(const float* __restrict__ tt,
                              const int* __restrict__ lens,
                              float* __restrict__ out)
{
    const int g = blockIdx.x * blockDim.x + threadIdx.x;
    if (g >= B_ * MTGT) return;
    const int b = g / MTGT, j = g % MTGT;
    const float* p = tt + (size_t)g * 6;
    const bool valid = j < lens[b];
    const float cx = p[0], cy = p[1], w = p[2], h = p[3], s = p[4];
    const float x1 = cx - 0.5f * w, y1 = cy - 0.5f * h;
    out[OTB + g * 4 + 0] = valid ? x1 : 0.f;
    out[OTB + g * 4 + 1] = valid ? y1 : 0.f;
    out[OTB + g * 4 + 2] = valid ? (x1 + w) : 0.f;
    out[OTB + g * 4 + 3] = valid ? (y1 + h) : 0.f;
    out[OTS + g] = valid ? s : 0.f;
    out[OTL + g] = valid ? (float)(int)p[5] : -1.f;
    out[OTV + g] = valid ? 1.f : 0.f;
}

extern "C" void kernel_launch(void* const* d_in, const int* in_sizes, int n_in,
                              void* d_out, int out_size, void* d_ws, size_t ws_size,
                              hipStream_t stream)
{
    (void)in_sizes; (void)n_in; (void)out_size; (void)ws_size;
    const float* preds = (const float*)d_in[0];
    const float* tt    = (const float*)d_in[1];
    const int*   lens  = (const int*)d_in[2];
    float* out = (float*)d_out;

    char* ws = (char*)d_ws;
    float* scores  = (float*)ws;                                  // B*N floats
    int*   top_idx = (int*)(ws + (size_t)B_ * NPRED * 4);         // B*K ints
    float* top_s   = (float*)(ws + (size_t)B_ * NPRED * 4 + (size_t)B_ * KDET * 4);

    dim3 g1((NPRED + ROWS_PER_BLOCK - 1) / ROWS_PER_BLOCK, B_);
    score_kernel <<<g1, ROWS_PER_BLOCK, 0, stream>>>(preds, scores);
    topk_kernel  <<<B_, 1024, 0, stream>>>(scores, top_idx, top_s);
    nms_kernel   <<<B_, 320, 0, stream>>>(preds, top_idx, top_s, out);
    target_kernel<<<(B_ * MTGT + 255) / 256, 256, 0, stream>>>(tt, lens, out);
}
